// MoE_180388627385
// MI455X (gfx1250) — compile-verified, matrix operand order checked by reading, not run on hardware
//
#include <hip/hip_runtime.h>

#define N_TOKENS 32768
#define D_IN     512
#define D_OUT    512
#define N_EXPERTS 8

typedef __attribute__((ext_vector_type(8)))  __bf16 v8bf;
typedef __attribute__((ext_vector_type(16))) __bf16 v16bf;
typedef __attribute__((ext_vector_type(8)))  float  v8f;

// ---------------------------------------------------------------------------
// Kernel 0: zero per-expert counters; convert+swizzle W (fp32 -> bf16) into the
// exact per-lane B-operand layout of V_WMMA_F32_16X16X32_BF16:
//   tile = 32(K) x 16(N); lane = nn + 16*((kk>>3)&1); halfword = (kk&7)+8*(kk>>4)
// Packed tile = 32 lanes x 16 bf16 contiguous (1 KB) -> coalesced b128 loads.
// ---------------------------------------------------------------------------
__global__ __launch_bounds__(256)
void moe_pack_w(const float* __restrict__ W, __bf16* __restrict__ Wpack,
                int* __restrict__ counts) {
  if (blockIdx.x == 0 && threadIdx.x < N_EXPERTS) counts[threadIdx.x] = 0;
  const int id  = blockIdx.x * 256 + threadIdx.x;      // grid sized exactly
  const int e   = id / (D_IN * D_OUT);
  const int rem = id % (D_IN * D_OUT);
  const int k   = rem / D_OUT;
  const int n   = rem % D_OUT;
  const float v = W[id];                               // coalesced over n
  const int n0 = n >> 4, nn = n & 15;
  const int k0 = k >> 5, kk = k & 31;
  const int lane = nn + (((kk >> 3) & 1) << 4);
  const int hw   = (kk & 7) + ((kk >> 4) << 3);
  const size_t off = ((((size_t)(e * 32 + n0) * 16 + k0) * 32 + lane) * 16) + hw;
  Wpack[off] = (__bf16)v;
}

// ---------------------------------------------------------------------------
// Kernel 1: gating. One wave32 per token.
//  - per-lane partial dot over 16 dims for all 8 experts, shfl_xor reduction
//  - softmax, top-2, gates g = p/(p0+p1+1e-6)  (matches reference)
//  - lane 0 appends (token, gate) to per-expert lists via atomic counters
//  - all lanes write y = g0*b[e0] + g1*b[e1]   (serves as y initialization)
// ---------------------------------------------------------------------------
__global__ __launch_bounds__(256)
void moe_gate(const float* __restrict__ x, const float* __restrict__ wg,
              const float* __restrict__ b, float* __restrict__ y,
              int* __restrict__ counts, int* __restrict__ toklist,
              float* __restrict__ gatelist) {
  const int lane  = threadIdx.x & 31;
  const int token = blockIdx.x * 8 + (threadIdx.x >> 5);

  float acc[N_EXPERTS];
#pragma unroll
  for (int e = 0; e < N_EXPERTS; ++e) acc[e] = 0.f;

#pragma unroll
  for (int j = 0; j < 16; ++j) {
    const int d = lane + 32 * j;                       // coalesced 128B per j
    const float xv = x[(size_t)token * D_IN + d];
    const float* w = wg + d * N_EXPERTS;               // 32B row, contiguous
#pragma unroll
    for (int e = 0; e < N_EXPERTS; ++e) acc[e] += xv * w[e];
  }

#pragma unroll
  for (int off = 16; off > 0; off >>= 1)
#pragma unroll
    for (int e = 0; e < N_EXPERTS; ++e) acc[e] += __shfl_xor(acc[e], off, 32);

  // all lanes hold full logits now; redundant softmax avoids broadcasts
  float mx = acc[0];
#pragma unroll
  for (int e = 1; e < N_EXPERTS; ++e) mx = fmaxf(mx, acc[e]);
  float p[N_EXPERTS], den = 0.f;
#pragma unroll
  for (int e = 0; e < N_EXPERTS; ++e) { p[e] = __expf(acc[e] - mx); den += p[e]; }
  const float inv = 1.f / den;
#pragma unroll
  for (int e = 0; e < N_EXPERTS; ++e) p[e] *= inv;

  int e0 = 0;
#pragma unroll
  for (int e = 1; e < N_EXPERTS; ++e) if (p[e] > p[e0]) e0 = e;
  int e1 = (e0 == 0) ? 1 : 0;
#pragma unroll
  for (int e = 0; e < N_EXPERTS; ++e) if (e != e0 && p[e] > p[e1]) e1 = e;

  const float s  = p[e0] + p[e1] + 1e-6f;
  const float g0 = p[e0] / s;
  const float g1 = p[e1] / s;

  if (lane == 0) {
    int pos0 = atomicAdd(counts + e0, 1);
    toklist [e0 * N_TOKENS + pos0] = token;
    gatelist[e0 * N_TOKENS + pos0] = g0;
    int pos1 = atomicAdd(counts + e1, 1);
    toklist [e1 * N_TOKENS + pos1] = token;
    gatelist[e1 * N_TOKENS + pos1] = g1;
  }

  const float* b0 = b + e0 * D_OUT;
  const float* b1 = b + e1 * D_OUT;
#pragma unroll
  for (int j = 0; j < 16; ++j) {
    const int n = lane + 32 * j;
    y[(size_t)token * D_OUT + n] = g0 * b0[n] + g1 * b1[n];
  }
}

// ---------------------------------------------------------------------------
// Kernel 2: grouped GEMM per expert with WMMA bf16 -> f32.
// Block: 256 thr = 8 waves; tile 64 tokens x 128 outputs; K step 32.
// A: gathered fp32 x rows copied global->LDS with GLOBAL_LOAD_ASYNC_TO_LDS_B128
//    (ASYNCcnt-tracked, triple-buffered, one barrier per K step); converted to
//    bf16 per-lane while building the WMMA A fragment (v_cvt_pk_bf16_f32
//    co-executes with the XDL WMMA pipe).
// B: pre-swizzled bf16 panels from L2 (one v16bf = 32B per lane, coalesced).
// C: scaled by gate and atomically accumulated into y (2 experts per token).
// K loop fully unrolled: %3 buffer indices become constants, giving the
// scheduler distinct B registers per K-step for deep load/WMMA pipelining.
// ---------------------------------------------------------------------------
#define A_LD 36          // fp32 row stride in LDS (32 + pad): banks spread
#define KT_N 16          // 512 / 32 K-steps

__global__ __launch_bounds__(256)
void moe_gemm(const float* __restrict__ x, const __bf16* __restrict__ Wpack,
              const int* __restrict__ counts, const int* __restrict__ toklist,
              const float* __restrict__ gatelist, float* __restrict__ y) {
  const int e   = blockIdx.z;
  const int cnt = counts[e];
  const int m0  = blockIdx.x * 64;
  if (m0 >= cnt) return;                 // block-uniform: EXEC stays all-ones

  __shared__ float  Asf[3][64][A_LD];    // triple-buffered 64x32 fp32 A tile
  __shared__ int    tokS[64];
  __shared__ float  gateS[64];

  const int tid = threadIdx.x;
  if (tid < 64) {
    const int gm = m0 + tid;
    if (gm < cnt) {
      tokS [tid] = toklist [e * N_TOKENS + gm];
      gateS[tid] = gatelist[e * N_TOKENS + gm];
    } else {
      tokS[tid] = 0; gateS[tid] = 0.f;   // safe dummy row, masked at store
    }
  }
  __syncthreads();

  const int lane    = tid & 31;
  const int wid     = tid >> 5;
  const int waveM   = wid >> 1;          // 0..3 : 16-row slice
  const int waveN   = wid & 1;           // 0..1 : 64-col slice
  const int rowA    = (waveM << 4) + (lane & 15);
  const int halfSel = lane >> 4;         // A-fragment K-octet select

  const int ldrow  = tid >> 2;           // staging: row 0..63
  const int ldpart = tid & 3;            // staging: 8 floats (32B) each
  const float* xsrc = x + (size_t)tokS[ldrow] * D_IN + ldpart * 8;

  const int ntile0 = blockIdx.y * 8 + waveN * 4;
  const __bf16* bbase = Wpack + ((size_t)(e * 32 + ntile0) * 16) * 512 + lane * 16;

  v8f cacc[4];
  const v8f vzero = {0.f, 0.f, 0.f, 0.f, 0.f, 0.f, 0.f, 0.f};
#pragma unroll
  for (int i = 0; i < 4; ++i) cacc[i] = vzero;

  // prologue: async-copy tile 0 into buffer 0 (2 x b128 per thread)
  {
    unsigned lds0 = (unsigned)(uintptr_t)&Asf[0][ldrow][ldpart * 8];
    const float* g = xsrc;
    asm volatile(
        "global_load_async_to_lds_b128 %0, %1, off\n\t"
        "global_load_async_to_lds_b128 %0, %1, off offset:16"
        :: "v"(lds0), "v"(g) : "memory");
  }

#pragma unroll
  for (int kt = 0; kt < KT_N; ++kt) {
    // issue next tile into buffer (kt+1)%3 (same buffer as (kt-2)%3, quiescent)
    if (kt + 1 < KT_N) {
      unsigned lds1 = (unsigned)(uintptr_t)&Asf[(kt + 1) % 3][ldrow][ldpart * 8];
      const float* g = xsrc + (size_t)(kt + 1) * 32;
      asm volatile(
          "global_load_async_to_lds_b128 %0, %1, off\n\t"
          "global_load_async_to_lds_b128 %0, %1, off offset:16"
          :: "v"(lds1), "v"(g) : "memory");
      asm volatile("s_wait_asynccnt 0x2" ::: "memory");  // tile kt landed
    } else {
      asm volatile("s_wait_asynccnt 0x0" ::: "memory");
    }
    __syncthreads();                     // publish tile kt to all waves

    if (kt + 1 < KT_N)                   // global_prefetch_b8 of next B panel
      __builtin_prefetch((const void*)(bbase + (size_t)(kt + 1) * 512), 0, 1);

    // build bf16 A fragment from fp32 LDS tile (K octets per ISA layout)
    const float* arow = &Asf[kt % 3][rowA][0];
    const float4 qa = *(const float4*)(arow + halfSel * 8);
    const float4 qb = *(const float4*)(arow + halfSel * 8 + 4);
    const float4 qc = *(const float4*)(arow + 16 + halfSel * 8);
    const float4 qd = *(const float4*)(arow + 16 + halfSel * 8 + 4);
    v16bf a;
    a[0]  = (__bf16)qa.x; a[1]  = (__bf16)qa.y; a[2]  = (__bf16)qa.z; a[3]  = (__bf16)qa.w;
    a[4]  = (__bf16)qb.x; a[5]  = (__bf16)qb.y; a[6]  = (__bf16)qb.z; a[7]  = (__bf16)qb.w;
    a[8]  = (__bf16)qc.x; a[9]  = (__bf16)qc.y; a[10] = (__bf16)qc.z; a[11] = (__bf16)qc.w;
    a[12] = (__bf16)qd.x; a[13] = (__bf16)qd.y; a[14] = (__bf16)qd.z; a[15] = (__bf16)qd.w;

#pragma unroll
    for (int nt = 0; nt < 4; ++nt) {
      const v16bf bf = *(const v16bf*)(bbase + (size_t)nt * 16 * 512 + (size_t)kt * 512);
      cacc[nt] = __builtin_amdgcn_wmma_f32_16x16x32_bf16(
          /*neg_a=*/false, a, /*neg_b=*/false, bf,
          /*c_mod=*/(short)0, cacc[nt], /*reuse_a=*/false, /*reuse_b=*/false);
    }
  }

  // C layout: lane 0-15 -> M=vgpr r, lane 16-31 -> M=8+r; N = lane&15
  const int colBase = blockIdx.y * 128 + waveN * 64 + (lane & 15);
#pragma unroll
  for (int nt = 0; nt < 4; ++nt) {
#pragma unroll
    for (int r = 0; r < 8; ++r) {
      const int ml = (waveM << 4) + (halfSel << 3) + r;
      if (m0 + ml < cnt) {
        atomicAdd(y + (size_t)tokS[ml] * D_OUT + colBase + nt * 16,
                  gateS[ml] * cacc[nt][r]);
      }
    }
  }
}

// ---------------------------------------------------------------------------
extern "C" void kernel_launch(void* const* d_in, const int* in_sizes, int n_in,
                              void* d_out, int out_size, void* d_ws, size_t ws_size,
                              hipStream_t stream) {
  const float* x  = (const float*)d_in[0];   // (32768, 512)
  const float* wg = (const float*)d_in[1];   // (512, 8)
  const float* W  = (const float*)d_in[2];   // (8, 512, 512)
  const float* b  = (const float*)d_in[3];   // (8, 512)
  float* y = (float*)d_out;                  // (32768, 512)

  char* ws = (char*)d_ws;
  int*    counts   = (int*)(ws);                                       // 32 B
  int*    toklist  = (int*)(ws + 256);                                 // 1 MB
  float*  gatelist = (float*)(ws + 256 + (size_t)N_EXPERTS * N_TOKENS * 4);
  __bf16* wpack    = (__bf16*)(ws + 256 + 2 * (size_t)N_EXPERTS * N_TOKENS * 4); // 4 MB

  moe_pack_w<<<(N_EXPERTS * D_IN * D_OUT) / 256, 256, 0, stream>>>(W, wpack, counts);
  moe_gate<<<N_TOKENS / 8, 256, 0, stream>>>(x, wg, b, y, counts, toklist, gatelist);
  dim3 g2(N_TOKENS / 64, D_OUT / 128, N_EXPERTS);
  moe_gemm<<<g2, 256, 0, stream>>>(x, wpack, counts, toklist, gatelist, y);
}
